// LSTMCRBM_45999099740347
// MI455X (gfx1250) — compile-verified, matrix operand order checked by reading, not run on hardware
//
#include <hip/hip_runtime.h>
#include <hip/hip_bf16.h>
#include <stdint.h>

// Problem sizes (fixed by the reference)
#define B_  4096
#define T_  64
#define F_  128
#define U_  256
#define H_  256
#define G4  1024      // 4*U
#define NW  32768     // F*H
#define MC  32

typedef __attribute__((ext_vector_type(16))) __bf16        bf16x16;
typedef __attribute__((ext_vector_type(8)))  float         f32x8;
typedef __attribute__((ext_vector_type(4)))  unsigned int  u32x4;

union ABv { u32x4 u[2]; bf16x16 v; };

__device__ __forceinline__ float bf2f(unsigned short u) {
  return __uint_as_float(((unsigned int)u) << 16);
}
__device__ __forceinline__ unsigned short f2bf(float f) {
  unsigned int u = __float_as_uint(f);
  u += 0x7FFFu + ((u >> 16) & 1u);           // round-to-nearest-even
  return (unsigned short)(u >> 16);
}
__device__ __forceinline__ unsigned int pcg(unsigned int x) {
  x = x * 747796405u + 2891336453u;
  unsigned int w = ((x >> ((x >> 28) + 4u)) ^ x) * 277803737u;
  return (w >> 22) ^ w;
}
__device__ __forceinline__ float rng_uniform(unsigned int a, unsigned int b, unsigned int c) {
  unsigned int s = pcg(a ^ 0x9E3779B1u);
  s = pcg(s + b * 0x85EBCA77u);
  s = pcg(s + c * 0xC2B2AE3Du);
  return ((float)s + 0.5f) * 2.3283064e-10f;  // (0,1)
}
__device__ __forceinline__ float rng_normal(unsigned int a, unsigned int b, unsigned int c) {
  float u1 = rng_uniform(a, b, 2u * c);
  float u2 = rng_uniform(a, b, 2u * c + 1u);
  float r = sqrtf(-2.0f * logf(u1));
  return r * cosf(6.28318530718f * u2);
}
__device__ __forceinline__ float sigmoidf_(float x) { return 1.0f / (1.0f + __expf(-x)); }

// ---------------------------------------------------------------- utilities
__global__ void k_f32_to_bf16(const float* __restrict__ in,
                              unsigned short* __restrict__ out, int n) {
  int i = blockIdx.x * blockDim.x + threadIdx.x;
  int stride = gridDim.x * blockDim.x;
  for (; i < n; i += stride) out[i] = f2bf(in[i]);
}

__global__ void k_add_bias(const float* __restrict__ a, const float* __restrict__ b,
                           float* __restrict__ out, int n) {
  int i = blockIdx.x * blockDim.x + threadIdx.x;
  if (i < n) out[i] = a[i] + b[i];
}

// ---------------------------------------------------------------- LSTM
// One block = 16 batch rows, all 64 timesteps. 8 waves; wave w owns gate
// columns u in [32w, 32w+32) for all four gates (i,f,g,o) so the gate
// nonlinearity is pure register math on matching WMMA accumulator slots.
// Weights stream from L2 every step; an opaque per-iteration zero keeps the
// compiler from hoisting (and spilling) the 96 B-tiles across timesteps.
__global__ void __launch_bounds__(256) k_lstm(
    const unsigned short* __restrict__ x_bf, // [B,T,F] bf16 (pre-converted)
    const unsigned short* __restrict__ Wih,  // [G4,F]  bf16 row-major
    const unsigned short* __restrict__ Whh,  // [G4,U]  bf16 row-major
    const float* __restrict__ bias,          // [G4] = b_ih + b_hh
    float* __restrict__ xh,                  // [B,U] f32 out
    unsigned short* __restrict__ xh_bf)      // [B,U] bf16 out
{
  __shared__ unsigned short sX[16 * F_];   // x_t tile (bf16)      4 KB
  __shared__ unsigned short sH[16 * U_];   // h tile   (bf16)      8 KB
  __shared__ float sB[G4];                 // combined gate bias   4 KB

  const int tid  = threadIdx.x;
  const int lane = tid & 31;
  const int wv   = tid >> 5;               // wave 0..7
  const int b0   = blockIdx.x * 16;
  const int col  = lane & 15;              // M-row (A) / N-col (B/C) selector
  const int hiK  = (lane >> 4) * 16;       // K-half split across lane groups

  for (int i = tid; i < G4; i += 256)      sB[i] = bias[i];
  for (int i = tid; i < 16 * U_; i += 256) sH[i] = 0;

  // per-thread staging geometry for the async x_t copy (16 B / lane / step)
  const int sidx = tid * 8;                        // half index in tile
  const int sm = sidx >> 7, scc = sidx & 127;
  const unsigned int sX_lds = (unsigned int)(uintptr_t)(&sX[sidx]);

  float c0[2][8];
  #pragma unroll
  for (int j = 0; j < 2; ++j)
    #pragma unroll
    for (int r = 0; r < 8; ++r) c0[j][r] = 0.f;

  #pragma unroll 1
  for (int t = 0; t < T_; ++t) {
    __syncthreads();                       // gate-phase sH writes visible

    // ---- stage x_t tile via CDNA5 async copy (ASYNCcnt-tracked)
    {
      const unsigned short* gsrc =
          x_bf + (((size_t)(b0 + sm) * T_ + t) * F_ + scc);
      asm volatile("global_load_async_to_lds_b128 %0, %1, off"
                   :: "v"(sX_lds), "v"(gsrc) : "memory");
      asm volatile("s_wait_asynccnt 0x0" ::: "memory");
    }
    __syncthreads();

    // opaque zero: makes weight addresses loop-variant -> no LICM, no spills
    unsigned int tz;
    asm volatile("v_mov_b32 %0, 0" : "=v"(tz));

    f32x8 acc[4][2] = {};                  // [gate][j] accumulators

    // ---- G += x_t @ Wih^T   (K = F = 128, 4 K-blocks)
    #pragma unroll
    for (int kb = 0; kb < 4; ++kb) {
      ABv A;
      const u32x4* ap = (const u32x4*)(&sX[col * F_ + kb * 32 + hiK]);
      A.u[0] = ap[0]; A.u[1] = ap[1];
      #pragma unroll
      for (int g = 0; g < 4; ++g)
        #pragma unroll
        for (int j = 0; j < 2; ++j) {
          int row = g * U_ + (2 * wv + j) * 16 + col;   // W row = gate output
          const u32x4* bp = (const u32x4*)(
              Wih + (size_t)row * F_ + (unsigned)(kb * 32 + hiK) + tz);
          ABv Bv; Bv.u[0] = bp[0]; Bv.u[1] = bp[1];
          acc[g][j] = __builtin_amdgcn_wmma_f32_16x16x32_bf16(
              false, A.v, false, Bv.v, (short)0, acc[g][j], false, false);
        }
    }
    // ---- G += h @ Whh^T     (K = U = 256, 8 K-blocks)
    #pragma unroll
    for (int kb = 0; kb < 8; ++kb) {
      ABv A;
      const u32x4* ap = (const u32x4*)(&sH[col * U_ + kb * 32 + hiK]);
      A.u[0] = ap[0]; A.u[1] = ap[1];
      #pragma unroll
      for (int g = 0; g < 4; ++g)
        #pragma unroll
        for (int j = 0; j < 2; ++j) {
          int row = g * U_ + (2 * wv + j) * 16 + col;
          const u32x4* bp = (const u32x4*)(
              Whh + (size_t)row * U_ + (unsigned)(kb * 32 + hiK) + tz);
          ABv Bv; Bv.u[0] = bp[0]; Bv.u[1] = bp[1];
          acc[g][j] = __builtin_amdgcn_wmma_f32_16x16x32_bf16(
              false, A.v, false, Bv.v, (short)0, acc[g][j], false, false);
        }
    }

    __syncthreads();                       // all sH reads done before rewrite

    // ---- gates: all in registers (i,f,g,o share lane/slot layout)
    #pragma unroll
    for (int j = 0; j < 2; ++j) {
      int u = (2 * wv + j) * 16 + col;
      float bi = sB[u], bf = sB[U_ + u], bg = sB[2 * U_ + u], bo = sB[3 * U_ + u];
      #pragma unroll
      for (int r = 0; r < 8; ++r) {
        int m = r + ((lane >> 4) << 3);    // C layout: lanes 16..31 -> M+8
        float iv = sigmoidf_(acc[0][j][r] + bi);
        float fv = sigmoidf_(acc[1][j][r] + bf);
        float gv = tanhf   (acc[2][j][r] + bg);
        float ov = sigmoidf_(acc[3][j][r] + bo);
        float cn = fv * c0[j][r] + iv * gv;
        c0[j][r] = cn;
        float hn = ov * tanhf(cn);
        sH[m * U_ + u] = f2bf(hn);
        if (t == T_ - 1) {
          xh   [(size_t)(b0 + m) * U_ + u] = hn;
          xh_bf[(size_t)(b0 + m) * U_ + u] = f2bf(hn);
        }
      }
    }
  }
}

// ---------------------------------------------------------------- weights GEMM
// Wout[b, f*H+h] = (xh @ wts_W^T + wts_b) as bf16.  M=4096, N=32768, K=256.
__global__ void __launch_bounds__(256) k_wgemm(
    const unsigned short* __restrict__ xh_bf,  // [B,U] bf16
    const unsigned short* __restrict__ Wts,    // [NW,U] bf16 row-major
    const float* __restrict__ wts_b,           // [NW]
    unsigned short* __restrict__ Wout)         // [B,NW] bf16
{
  __shared__ unsigned short sA[16 * U_];       // 8 KB A tile
  const int tid = threadIdx.x, lane = tid & 31, wv = tid >> 5;
  const int col = lane & 15, hiK = (lane >> 4) * 16;
  const int m0 = blockIdx.x * 16;
  const int nsup = blockIdx.y;                 // 0..31, 1024 N-cols each

  {
    const u32x4* src = (const u32x4*)(xh_bf + (size_t)m0 * U_);
    u32x4* dst = (u32x4*)sA;                   // 512 u32x4 total
    dst[tid]       = src[tid];
    dst[tid + 256] = src[tid + 256];
  }
  __syncthreads();

  f32x8 acc[8] = {};
  #pragma unroll
  for (int kb = 0; kb < 8; ++kb) {
    ABv A;
    const u32x4* ap = (const u32x4*)(&sA[col * U_ + kb * 32 + hiK]);
    A.u[0] = ap[0]; A.u[1] = ap[1];
    #pragma unroll
    for (int nt = 0; nt < 8; ++nt) {
      int ntile = nsup * 64 + wv * 8 + nt;
      int row = ntile * 16 + col;
      const unsigned short* bptr = Wts + (size_t)row * U_ + kb * 32 + hiK;
      if (kb < 7 && nt == 0) __builtin_prefetch(bptr + 32, 0, 0);  // next K-block
      const u32x4* bp = (const u32x4*)bptr;
      ABv Bv; Bv.u[0] = bp[0]; Bv.u[1] = bp[1];
      acc[nt] = __builtin_amdgcn_wmma_f32_16x16x32_bf16(
          false, A.v, false, Bv.v, (short)0, acc[nt], false, false);
    }
  }
  #pragma unroll
  for (int nt = 0; nt < 8; ++nt) {
    int ncol = (nsup * 64 + wv * 8 + nt) * 16 + col;
    float bb = wts_b[ncol];
    int mbase = m0 + ((lane >> 4) << 3);
    #pragma unroll
    for (int r = 0; r < 8; ++r)
      Wout[(size_t)(mbase + r) * NW + ncol] = f2bf(acc[nt][r] + bb);
  }
}

// ---------------------------------------------------------------- Gibbs
// One block per sample. Per-sample weights (64 KB bf16) resident in LDS for
// all 32 steps; LDS layout swizzled (h -> (h+f)&255) so both h-major and
// f-major dot products are bank-conflict-light.
__global__ void __launch_bounds__(256) k_gibbs(
    const float* __restrict__ xh,              // [B,U]
    const unsigned short* __restrict__ Wall,   // [B,NW] bf16
    const float* __restrict__ bias_W, const float* __restrict__ bias_b,
    const float* __restrict__ prec_W, const float* __restrict__ prec_b,
    float* __restrict__ out)                   // [B,F]
{
  extern __shared__ unsigned char smraw[];
  unsigned short* sW   = (unsigned short*)smraw;        // NW halfs (swizzled)
  float* sxh  = (float*)(smraw + NW * 2);               // U
  float* sbias = sxh + U_;                              // F
  float* sprec = sbias + F_;
  float* sstd  = sprec + F_;
  float* sy    = sstd  + F_;
  float* smean = sy    + F_;
  float* shv   = smean + F_;                            // H

  const int tid = threadIdx.x;
  const unsigned int b = blockIdx.x;

  sxh[tid] = xh[(size_t)b * U_ + tid];

  { // load + swizzle per-sample weights: W[f][h] -> sW[f*H + ((h+f)&255)]
    const u32x4* src = (const u32x4*)(Wall + (size_t)b * NW);
    #pragma unroll 4
    for (int k = 0; k < 16; ++k) {
      int unit = tid + k * 256;                 // 4096 u32x4 total
      u32x4 v = src[unit];
      int f  = (unit * 8) >> 8;
      int h0 = (unit * 8) & 255;
      const unsigned short* pv = (const unsigned short*)&v;
      #pragma unroll
      for (int e = 0; e < 8; ++e)
        sW[f * H_ + ((h0 + e + f) & 255)] = pv[e];
    }
  }
  __syncthreads();

  if (tid < F_) {   // conditioning nets: bias / precision / std / y0
    const int f = tid;
    float a0 = bias_b[f], a1 = prec_b[f];
    const float* bw = bias_W + (size_t)f * U_;
    const float* pw = prec_W + (size_t)f * U_;
    #pragma unroll 4
    for (int u = 0; u < U_; ++u) { float xv = sxh[u]; a0 += bw[u] * xv; a1 += pw[u] * xv; }
    float sp = (a1 > 20.f) ? a1 : ((a1 < -20.f) ? __expf(a1) : log1pf(__expf(a1)));
    float pr = sp + 1e-6f;
    float sd = rsqrtf(pr);
    sbias[f] = a0; sprec[f] = pr; sstd[f] = sd;
    sy[f] = a0 + sd * rng_normal(b, 0xFFFFu, (unsigned)f);
  }
  __syncthreads();

  for (int step = 0; step < MC; ++step) {
    { // hidden units: h_prob = sigmoid(y . W[:,h]); sample bernoulli
      const int h = tid;
      float acc = 0.f;
      #pragma unroll 4
      for (int f = 0; f < F_; ++f)
        acc += sy[f] * bf2f(sW[f * H_ + ((h + f) & 255)]);
      float p = sigmoidf_(acc);
      shv[h] = (rng_uniform(b, 2u * step, (unsigned)h) < p) ? 1.f : 0.f;
    }
    __syncthreads();
    if (tid < F_) { // visible: mean = bias + (W h)/prec; y = mean + std*noise
      const int f = tid;
      float acc = 0.f;
      #pragma unroll 4
      for (int h = 0; h < H_; ++h)
        acc += bf2f(sW[f * H_ + ((h + f) & 255)]) * shv[h];
      float mean = sbias[f] + acc / sprec[f];
      smean[f] = mean;
      sy[f] = mean + sstd[f] * rng_normal(b, 2u * step + 1u, (unsigned)f);
    }
    __syncthreads();
  }
  if (tid < F_) out[(size_t)b * F_ + tid] = smean[tid];   // denoise: last mean
}

// ---------------------------------------------------------------- launch
extern "C" void kernel_launch(void* const* d_in, const int* in_sizes, int n_in,
                              void* d_out, int out_size, void* d_ws, size_t ws_size,
                              hipStream_t stream) {
  (void)in_sizes; (void)n_in; (void)out_size; (void)ws_size;
  const float* x      = (const float*)d_in[0];
  const float* W_ih   = (const float*)d_in[1];
  const float* W_hh   = (const float*)d_in[2];
  const float* b_ih   = (const float*)d_in[3];
  const float* b_hh   = (const float*)d_in[4];
  const float* bias_W = (const float*)d_in[5];
  const float* bias_b = (const float*)d_in[6];
  const float* prec_W = (const float*)d_in[7];
  const float* prec_b = (const float*)d_in[8];
  const float* wts_W  = (const float*)d_in[9];
  const float* wts_b  = (const float*)d_in[10];
  float* out = (float*)d_out;

  char* ws = (char*)d_ws;
  unsigned short* Wih_bf = (unsigned short*)(ws + 0);            // 256 KB
  unsigned short* Whh_bf = (unsigned short*)(ws + 262144);       // 512 KB
  float*          biasL  = (float*)        (ws + 786432);        //   4 KB
  unsigned short* wts_bf = (unsigned short*)(ws + 790528);       //  16 MB
  float*          xh     = (float*)        (ws + 17567744);      //   4 MB
  unsigned short* xh_bf  = (unsigned short*)(ws + 21762048);     //   2 MB
  unsigned short* x_bf   = (unsigned short*)(ws + 23859200);     //  64 MB
  unsigned short* Wall   = (unsigned short*)(ws + 90968064);     // 256 MB

  k_f32_to_bf16<<<256,  256, 0, stream>>>(W_ih,  Wih_bf, G4 * F_);
  k_f32_to_bf16<<<512,  256, 0, stream>>>(W_hh,  Whh_bf, G4 * U_);
  k_f32_to_bf16<<<4096, 256, 0, stream>>>(wts_W, wts_bf, NW * U_);
  k_f32_to_bf16<<<8192, 256, 0, stream>>>(x,     x_bf,   B_ * T_ * F_);
  k_add_bias  <<<4,    256, 0, stream>>>(b_ih, b_hh, biasL, G4);

  k_lstm <<<B_ / 16, 256, 0, stream>>>(x_bf, Wih_bf, Whh_bf, biasL, xh, xh_bf);
  k_wgemm<<<dim3(B_ / 16, NW / 1024), 256, 0, stream>>>(xh_bf, wts_bf, wts_b, Wall);

  const size_t gibbs_lds = (size_t)NW * 2 + U_ * 4 + F_ * 4 * 5 + H_ * 4;  // ~68.5 KB
  hipFuncSetAttribute((const void*)k_gibbs,
                      hipFuncAttributeMaxDynamicSharedMemorySize, (int)gibbs_lds);
  k_gibbs<<<B_, 256, gibbs_lds, stream>>>(xh, Wall, bias_W, bias_b, prec_W, prec_b, out);
}
